// ScaledDotProdAtt_8667244003892
// MI455X (gfx1250) — matrix operand order repaired and checked by reading, not compiled, not run on hardware
//
#include <hip/hip_runtime.h>

typedef __attribute__((ext_vector_type(16))) _Float16 v16h;
typedef __attribute__((ext_vector_type(8)))  _Float16 v8h;
typedef __attribute__((ext_vector_type(4)))  _Float16 v4h;
typedef __attribute__((ext_vector_type(8)))  float    v8f;
typedef __attribute__((ext_vector_type(4)))  float    v4f;

#define B_    4
#define H_    16
#define S_    2048
#define DK_   64
#define QTILE 256   // query rows per block (8 waves x 2 subgroups x 16 rows)
#define KTILE 64    // key rows per tile
#define LDP   72    // padded LDS row stride in halves (144B: 16B-aligned, bank-spread)

__device__ __forceinline__ v16h cat8(v8h a, v8h b) {
  v16h r;
#pragma unroll
  for (int i = 0; i < 8; ++i) { r[i] = a[i]; r[8 + i] = b[i]; }
  return r;
}

__device__ __forceinline__ v8f zero8() {
  v8f z;
#pragma unroll
  for (int i = 0; i < 8; ++i) z[i] = 0.0f;
  return z;
}

__global__ __launch_bounds__(256)
void fattn_fwd_kernel(const float* __restrict__ q, const float* __restrict__ k,
                      const float* __restrict__ v, const int* __restrict__ sen_len,
                      float* __restrict__ out)
{
  __shared__ _Float16 Ks[KTILE * LDP];      // K tile, row-major [key][d]
  __shared__ _Float16 Vt[DK_ * LDP];        // V tile, transposed [d][key]
  __shared__ _Float16 Pq[8 * 32 * LDP];     // per-wave P scratch, [q][key] (32 q rows/wave)

  const int bh   = blockIdx.x;              // fused (b,h)
  const int b    = bh / H_;
  const int q0   = blockIdx.y * QTILE;
  const int slen = sen_len[b];

  const int tid  = threadIdx.x;
  const int lane = tid & 31;
  const int wave = tid >> 5;
  const int lrow = lane & 15;
  const int hi   = lane >> 4;

  const size_t head = (size_t)bh * S_ * DK_;
  const float* qh = q + head;
  const float* kh = k + head;
  const float* vh = v + head;
  float*       oh = out + head;

  const int wq0  = q0 + wave * 32;                          // wave's first query row
  const int wq0u = __builtin_amdgcn_readfirstlane(wq0);     // wave-uniform copy for s_cbranch

  // ---- Q^T B-fragments (dk x q) for both 16-row subgroups; scale*log2(e) folded in ----
  const float SCALE = 0.125f * 1.44269504088896340736f;     // 1/sqrt(64) * log2(e)
  v16h Qb[2][2];
  int   kmax8[2];
  float m_run[2], l_run[2];
  v8f   Oc[2][4];
#pragma unroll
  for (int s = 0; s < 2; ++s) {
    const int qrow = wq0 + s * 16 + lrow;
    const float* qr = qh + (size_t)qrow * DK_ + hi * 16;
#pragma unroll
    for (int w = 0; w < 2; ++w)
#pragma unroll
      for (int g = 0; g < 4; ++g) {
        v4f f = *(const v4f*)(qr + w * 32 + g * 4);
#pragma unroll
        for (int t = 0; t < 4; ++t) Qb[s][w][g * 4 + t] = (_Float16)(f[t] * SCALE);
      }
    const int kend = (slen - 1 < qrow) ? (slen - 1) : qrow; // keys valid iff key <= kend
    kmax8[s] = kend - 8 * hi;                               // fold C-layout row offset
    m_run[s] = -1e30f;
    l_run[s] = 0.0f;
#pragma unroll
    for (int dc = 0; dc < 4; ++dc) Oc[s][dc] = zero8();
  }

  const int jend = (q0 + QTILE < S_) ? (q0 + QTILE) : S_;   // causal bound for whole block
  _Float16* Pw = &Pq[wave * 32 * LDP];

  for (int j0 = 0; j0 < jend; j0 += KTILE) {
    __syncthreads();
    // ---- cooperative stage: K row-major (fp32->f16) ----
    {
      const int r  = tid >> 2;             // 0..63 key row
      const int cg = (tid & 3) * 16;       // d column group
      const float* kr = kh + (size_t)(j0 + r) * DK_ + cg;
      if (j0 + KTILE < jend) __builtin_prefetch(kr + (size_t)KTILE * DK_, 0, 3);
#pragma unroll
      for (int i = 0; i < 16; i += 4) {
        v4f kv = *(const v4f*)(kr + i);
#pragma unroll
        for (int t = 0; t < 4; ++t) Ks[r * LDP + cg + i + t] = (_Float16)kv[t];
      }
    }
    // ---- cooperative stage: V transposed via 4x4 register transpose (b64 stores) ----
    {
      const int r0 = (tid >> 4) * 4;       // key rows r0..r0+3
      const int c0 = (tid & 15) * 4;       // d cols  c0..c0+3
      const float* vr = vh + (size_t)(j0 + r0) * DK_ + c0;
      if (j0 + KTILE < jend) __builtin_prefetch(vr + (size_t)KTILE * DK_, 0, 3);
      v4f row[4];
#pragma unroll
      for (int j = 0; j < 4; ++j) row[j] = *(const v4f*)(vr + (size_t)j * DK_);
#pragma unroll
      for (int i = 0; i < 4; ++i) {
        v4h tp;
#pragma unroll
        for (int j = 0; j < 4; ++j) tp[j] = (_Float16)row[j][i];
        *(v4h*)(&Vt[(c0 + i) * LDP + r0]) = tp;
      }
    }
    __syncthreads();

    // ---- S^T = K Q^T : 4 key chunks, shared K A-frags, 2 q subgroups ----
    v8f Sc[2][4];
#pragma unroll
    for (int c = 0; c < 4; ++c) {
      const _Float16* ka = &Ks[(c * 16 + lrow) * LDP + hi * 16];
      v16h Ka0 = cat8(*(const v8h*)(ka),      *(const v8h*)(ka + 8));
      v16h Ka1 = cat8(*(const v8h*)(ka + 32), *(const v8h*)(ka + 40));
#pragma unroll
      for (int s = 0; s < 2; ++s) {
        v8f acc = zero8();
        acc = __builtin_amdgcn_wmma_f32_16x16x32_f16(false, Ka0, false, Qb[s][0], (short)0, acc, false, false);
        acc = __builtin_amdgcn_wmma_f32_16x16x32_f16(false, Ka1, false, Qb[s][1], (short)0, acc, false, false);
        Sc[s][c] = acc;
      }
    }

    // ---- mask only on boundary tiles (wave-uniform s_cbranch) ----
    // element (s,c,r): key = j0 + c*16 + 8*hi + r ; masked iff key > min(qrow_s, slen-1)
    if (j0 + KTILE - 1 > wq0u || j0 + KTILE > slen) {
#pragma unroll
      for (int s = 0; s < 2; ++s) {
        const int tloc = kmax8[s] - j0;
#pragma unroll
        for (int c = 0; c < 4; ++c)
#pragma unroll
          for (int r = 0; r < 8; ++r)
            if (c * 16 + r > tloc) Sc[s][c][r] = -1e9f;
      }
    }

    // ---- per-lane online softmax per subgroup; raw v_exp_f32 ----
#pragma unroll
    for (int s = 0; s < 2; ++s) {
      float mx = Sc[s][0][0];
#pragma unroll
      for (int c = 0; c < 4; ++c)
#pragma unroll
        for (int r = 0; r < 8; ++r)
          if (c | r) mx = fmaxf(mx, Sc[s][c][r]);
      mx = fmaxf(mx, __shfl_xor(mx, 16, 32));
      const float mnew  = fmaxf(m_run[s], mx);
      const float alpha = __builtin_amdgcn_exp2f(m_run[s] - mnew);
      m_run[s] = mnew;
      float rs = 0.0f;
#pragma unroll
      for (int c = 0; c < 4; ++c)
#pragma unroll
        for (int r = 0; r < 8; ++r) {
          const float p = __builtin_amdgcn_exp2f(Sc[s][c][r] - mnew);
          Sc[s][c][r] = p;
          rs += p;
        }
      rs += __shfl_xor(rs, 16, 32);
      l_run[s] = l_run[s] * alpha + rs;
#pragma unroll
      for (int dc = 0; dc < 4; ++dc)
#pragma unroll
        for (int r = 0; r < 8; ++r)
          Oc[s][dc][r] *= alpha;

      // P -> LDS [q][key]: 8 consecutive halves per chunk => ds_store_b128
#pragma unroll
      for (int c = 0; c < 4; ++c) {
        v8h pv;
#pragma unroll
        for (int r = 0; r < 8; ++r) pv[r] = (_Float16)Sc[s][c][r];
        *(v8h*)(&Pw[(s * 16 + lrow) * LDP + c * 16 + 8 * hi]) = pv;
      }
    }

    // ---- P^T B-fragments: lane col q = lrow, K = key = 16*hi + i (+32) ----
    v16h Pb[2][2];
#pragma unroll
    for (int s = 0; s < 2; ++s)
#pragma unroll
      for (int w = 0; w < 2; ++w) {
        const _Float16* pr = &Pw[(s * 16 + lrow) * LDP + w * 32 + hi * 16];
        Pb[s][w] = cat8(*(const v8h*)(pr), *(const v8h*)(pr + 8));
      }

    // ---- O^T += V^T P^T : 4 d-chunks, shared V A-frags, 2 q subgroups ----
#pragma unroll
    for (int dc = 0; dc < 4; ++dc) {
      const _Float16* va = &Vt[(dc * 16 + lrow) * LDP + hi * 16];
      v16h Va0 = cat8(*(const v8h*)(va),      *(const v8h*)(va + 8));
      v16h Va1 = cat8(*(const v8h*)(va + 32), *(const v8h*)(va + 40));
#pragma unroll
      for (int s = 0; s < 2; ++s) {
        Oc[s][dc] = __builtin_amdgcn_wmma_f32_16x16x32_f16(false, Va0, false, Pb[s][0], (short)0, Oc[s][dc], false, false);
        Oc[s][dc] = __builtin_amdgcn_wmma_f32_16x16x32_f16(false, Va1, false, Pb[s][1], (short)0, Oc[s][dc], false, false);
      }
    }
  }

  // ---- epilogue: O^T element (d = dc*16 + 8*hi + r, q = qrow_s) -> vector stores ----
#pragma unroll
  for (int s = 0; s < 2; ++s) {
    const float inv = 1.0f / l_run[s];
    float* orow = oh + (size_t)(wq0 + s * 16 + lrow) * DK_ + 8 * hi;
#pragma unroll
    for (int dc = 0; dc < 4; ++dc) {
      v4f o0, o1;
#pragma unroll
      for (int t = 0; t < 4; ++t) { o0[t] = Oc[s][dc][t] * inv; o1[t] = Oc[s][dc][4 + t] * inv; }
      *(v4f*)(orow + dc * 16)     = o0;
      *(v4f*)(orow + dc * 16 + 4) = o1;
    }
  }
}

extern "C" void kernel_launch(void* const* d_in, const int* in_sizes, int n_in,
                              void* d_out, int out_size, void* d_ws, size_t ws_size,
                              hipStream_t stream) {
  const float* q  = (const float*)d_in[0];
  const float* k  = (const float*)d_in[1];
  const float* v  = (const float*)d_in[2];
  const int*   sl = (const int*)d_in[3];
  float*       o  = (float*)d_out;
  (void)in_sizes; (void)n_in; (void)out_size; (void)d_ws; (void)ws_size;

  dim3 grid(B_ * H_, S_ / QTILE);   // (64, 8)
  fattn_fwd_kernel<<<grid, 256, 0, stream>>>(q, k, v, sl, o);
}
